// Model_41695542509986
// MI455X (gfx1250) — matrix operand order repaired and checked by reading, not compiled, not run on hardware
//
#include <hip/hip_runtime.h>
#include <hip/hip_bf16.h>
#include <math.h>

// Problem constants (from reference)
#define BB 64
#define SS 512
#define DD 768
#define LL 50
#define LP 64          // L padded to 4 WMMA n-tiles

typedef __attribute__((ext_vector_type(16))) __bf16 v16bf;
typedef __attribute__((ext_vector_type(8)))  float  v8f;
typedef __attribute__((ext_vector_type(4)))  int    v4i;

#if defined(__has_builtin)
#if __has_builtin(__builtin_amdgcn_global_load_async_to_lds_b128)
#define USE_ASYNC_LDS 1
#endif
#if __has_builtin(__builtin_amdgcn_s_wait_asynccnt)
#define HAVE_WAIT_ASYNC_BUILTIN 1
#endif
#endif

// ---------------------------------------------------------------- zero loss
__global__ void k_zero(float* out) {
    if (threadIdx.x == 0) out[0] = 0.0f;
}

// ------------------------------------------------- per-batch compaction scan
__global__ void k_scan(const int* __restrict__ valid_ids, int* __restrict__ srcIdx) {
    __shared__ int sc[SS];
    const int b = blockIdx.x;
    const int s = threadIdx.x;
    const int tok = b * SS + s;
    const int v = (valid_ids[tok] == 1) ? 1 : 0;
    srcIdx[tok] = -1;
    sc[s] = v;
    __syncthreads();
    for (int off = 1; off < SS; off <<= 1) {
        int t = (s >= off) ? sc[s - off] : 0;
        __syncthreads();
        sc[s] += t;
        __syncthreads();
    }
    if (v) srcIdx[b * SS + (sc[s] - 1)] = tok;
}

// ------------------- W (L x D fp32) -> bf16 padded to LP rows, + zero row
__global__ void k_wconv(const float* __restrict__ W, __bf16* __restrict__ Wb,
                        float* __restrict__ zrow) {
    const int i = blockIdx.x * blockDim.x + threadIdx.x;  // < LP*DD
    if (i < DD) zrow[i] = 0.0f;                           // dummy row for invalid tokens
    if (i >= LP * DD) return;
    const int n = i / DD;
    Wb[i] = (__bf16)(n < LL ? W[i] : 0.0f);               // hardware RNE convert
}

// ------------------------------------------------------------- WMMA GEMM
// Block: 256 threads = 8 waves. Wave -> 16 tokens x 64 cols (4 n-tiles).
// Block -> 128 tokens. Grid: 32768/128 = 256 blocks.
// Entire padded W^T (96 KB bf16) staged ONCE into dynamic LDS via async DMA;
// after one wait+barrier, 24 K-chunks run with no further synchronization.
__global__ void __launch_bounds__(256)
k_gemm(const float* __restrict__ seq,
       const int*   __restrict__ srcIdx,
       const __bf16* __restrict__ Wb,
       const float* __restrict__ zrow,
       const float* __restrict__ bias,
       float* __restrict__ logits) {
    extern __shared__ __bf16 ldsW[];   // LP*DD bf16 = 96 KB (layout == Wb)

    const int lane = threadIdx.x & 31;
    const int wave = threadIdx.x >> 5;
    const int tokenBase = blockIdx.x * 128 + wave * 16;
    const int m    = lane & 15;        // A row / C col index
    const int half = lane >> 4;        // K-half selector per ISA layout

    const int src = srcIdx[tokenBase + m];
    // Branch-free A source: invalid tokens read a zeroed dummy row.
    const float* row = (src >= 0) ? (seq + (size_t)src * DD) : zrow;

    // Warm the A stream while the W DMA is in flight.
    __builtin_prefetch(row, 0, 0);
    __builtin_prefetch(row + 256, 0, 0);
    __builtin_prefetch(row + 512, 0, 0);

    // ---- stage ALL of W^T: 6144 x b128 across 256 threads (24 each) ----
#ifdef USE_ASYNC_LDS
    #pragma unroll
    for (int i = 0; i < 24; ++i) {
        const int e = (threadIdx.x + i * 256) * 8;   // flat bf16 element offset
        __builtin_amdgcn_global_load_async_to_lds_b128(
            (v4i*)&Wb[e], (v4i*)&ldsW[e], 0, 0);
    }
#ifdef HAVE_WAIT_ASYNC_BUILTIN
    __builtin_amdgcn_s_wait_asynccnt(0);
#else
    asm volatile("s_wait_asynccnt 0x0" ::: "memory");
#endif
#else
    for (int i = 0; i < 24; ++i) {
        const int e = (threadIdx.x + i * 256) * 8;
        *(uint4*)&ldsW[e] = *(const uint4*)&Wb[e];
    }
#endif
    __syncthreads();   // all waves' W portions visible

    v8f c[4];
    #pragma unroll
    for (int t = 0; t < 4; ++t)
        c[t] = (v8f){0.f, 0.f, 0.f, 0.f, 0.f, 0.f, 0.f, 0.f};

    #pragma unroll
    for (int kc = 0; kc < DD; kc += 32) {
        // ---- A fragment: contiguous k = 16*half + s (matches B) ----
        const int kabs = kc + half * 16;
        float4 f0 = *(const float4*)(row + kabs);
        float4 f1 = *(const float4*)(row + kabs + 4);
        float4 f2 = *(const float4*)(row + kabs + 8);
        float4 f3 = *(const float4*)(row + kabs + 12);
        v16bf a;
        a[0]  = (__bf16)f0.x; a[1]  = (__bf16)f0.y; a[2]  = (__bf16)f0.z; a[3]  = (__bf16)f0.w;
        a[4]  = (__bf16)f1.x; a[5]  = (__bf16)f1.y; a[6]  = (__bf16)f1.z; a[7]  = (__bf16)f1.w;
        a[8]  = (__bf16)f2.x; a[9]  = (__bf16)f2.y; a[10] = (__bf16)f2.z; a[11] = (__bf16)f2.w;
        a[12] = (__bf16)f3.x; a[13] = (__bf16)f3.y; a[14] = (__bf16)f3.z; a[15] = (__bf16)f3.w;

        // ---- load all 4 B fragments, then 4 back-to-back WMMAs ----
        v16bf bf[4];
        #pragma unroll
        for (int t = 0; t < 4; ++t)
            bf[t] = *(const v16bf*)&ldsW[(t * 16 + m) * DD + kc + half * 16];
        #pragma unroll
        for (int t = 0; t < 4; ++t)
            c[t] = __builtin_amdgcn_wmma_f32_16x16x32_bf16(
                       false, a, false, bf[t], (short)0, c[t], false, false);
    }

    // ---- epilogue: bias + store (C layout: VGPR v -> row v+8*half, col lane&15)
    #pragma unroll
    for (int t = 0; t < 4; ++t) {
        const int n = t * 16 + m;
        if (n < LL) {
            const float bv = bias[n];
            #pragma unroll
            for (int v = 0; v < 8; ++v) {
                const int tok = tokenBase + v + half * 8;
                logits[(size_t)tok * LL + n] = c[t][v] + bv;
            }
        }
    }
}

// ------------------------------------------- fused log-softmax / NLL / mean
__global__ void k_loss(const float* __restrict__ logits,
                       const int*   __restrict__ labels,
                       const int*   __restrict__ lmask,
                       float* __restrict__ out) {
    __shared__ float red[SS];
    const int b = blockIdx.x;
    const int s = threadIdx.x;
    const int tok = b * SS + s;
    const float* lr = logits + (size_t)tok * LL;

    float mx = -INFINITY;
    #pragma unroll 5
    for (int l = 0; l < LL; ++l) mx = fmaxf(mx, lr[l]);
    float se = 0.0f;
    #pragma unroll 5
    for (int l = 0; l < LL; ++l) se += __expf(lr[l] - mx);
    const int lab = labels[tok];
    const float logp = lr[lab] - mx - __logf(se);
    red[s] = -logp * (float)lmask[tok];
    __syncthreads();
    for (int off = SS / 2; off > 0; off >>= 1) {
        if (s < off) red[s] += red[s + off];
        __syncthreads();
    }
    if (s == 0) atomicAdd(out, red[0] * (1.0f / ((float)SS * (float)BB)));
}

extern "C" void kernel_launch(void* const* d_in, const int* in_sizes, int n_in,
                              void* d_out, int out_size, void* d_ws, size_t ws_size,
                              hipStream_t stream) {
    const float* seq    = (const float*)d_in[0];   // [B,S,D] f32
    const int*   valid  = (const int*)  d_in[1];   // [B,S]
    const int*   labels = (const int*)  d_in[2];   // [B,S]
    const int*   lmask  = (const int*)  d_in[3];   // [B,S]
    const float* W      = (const float*)d_in[4];   // [L,D] f32
    const float* bias   = (const float*)d_in[5];   // [L]  f32

    float* out    = (float*)d_out;                 // [0]=avg_loss, [1..]=logits
    float* logits = out + 1;

    char* ws = (char*)d_ws;
    int*    srcIdx = (int*)ws;                                    // 128 KB
    __bf16* Wb     = (__bf16*)(ws + (size_t)BB * SS * 4);         // 96 KB
    float*  zrow   = (float*)(ws + (size_t)BB * SS * 4 + (size_t)LP * DD * 2); // 3 KB

    const int ldsBytes = LP * DD * (int)sizeof(__bf16);           // 96 KB dynamic LDS
    // Host-side attribute (not a stream op; graph-capture safe, deterministic).
    (void)hipFuncSetAttribute((const void*)k_gemm,
                              hipFuncAttributeMaxDynamicSharedMemorySize, ldsBytes);

    k_zero <<<1, 32, 0, stream>>>(out);
    k_scan <<<BB, SS, 0, stream>>>(valid, srcIdx);
    k_wconv<<<(LP * DD + 255) / 256, 256, 0, stream>>>(W, Wb, zrow);
    k_gemm <<<(BB * SS) / 128, 256, ldsBytes, stream>>>(seq, srcIdx, Wb, zrow, bias, logits);
    k_loss <<<BB, SS, 0, stream>>>(logits, labels, lmask, out);
}
